// SocialInteractionEncoder_78237124264598
// MI455X (gfx1250) — matrix operand order repaired and checked by reading, not compiled, not run on hardware
//
#include <hip/hip_runtime.h>
#include <hip/hip_bf16.h>

typedef __attribute__((ext_vector_type(16))) _Float16 v16h;
typedef __attribute__((ext_vector_type(8)))  float    v8f;

#define BATCH 32
#define SEQ   8192
#define DIM   128
#define NT    64

union Frag  { v16h v; uint4 q[2]; _Float16 h[16]; };
union Half8 { uint4 q; _Float16 h[8]; };

#define WMMA_F16(A, Bf, C) \
  __builtin_amdgcn_wmma_f32_16x16x32_f16(false, (A), false, (Bf), (short)0, (C), false, false)

// ---------------------------------------------------------------- utilities
__global__ void zero_kernel(float* p, int n) {
  int i = blockIdx.x * 256 + threadIdx.x;
  if (i < n) p[i] = 0.f;
}

__global__ void cast_f16_kernel(const float* __restrict__ src, _Float16* __restrict__ dst, int n) {
  int i = blockIdx.x * 256 + threadIdx.x;
  if (i < n) dst[i] = (_Float16)src[i];
}

// Pack W[O][Cin][3] (f32) into WMMA A-fragment order, K = tap*Cin + chan.
// dst index = ((mtile*KC + kc)*32 + lane)*16 + h
// A order used here (contiguous k, B matches): M = lane%16; K%32 = 16*(lane/16) + h
__global__ void pack_w_kernel(const float* __restrict__ W, _Float16* __restrict__ dst,
                              int O, int Cin) {
  int KK = Cin * 3;
  int total = O * KK;
  int idx = blockIdx.x * 256 + threadIdx.x;
  if (idx >= total) return;
  int h    = idx & 15;
  int lane = (idx >> 4) & 31;
  int ckc  = idx >> 9;
  int KC   = KK >> 5;
  int mt   = ckc / KC;
  int kc   = ckc % KC;
  int kp   = ((lane >> 4) << 4) + h;
  int K    = kc * 32 + kp;
  int M    = mt * 16 + (lane & 15);
  int t    = K / Cin;
  int i    = K % Cin;
  dst[idx] = (_Float16)W[(M * Cin + i) * 3 + t];
}

// ---------------------------------------------------------------- conv1: gather+GEMM
// O=256, Cin=128(embed), writes y1 f16 [B*S][256]
__global__ __launch_bounds__(256) void conv1_kernel(const int* __restrict__ ids,
                                                    const _Float16* __restrict__ embF,
                                                    const _Float16* __restrict__ W1p,
                                                    _Float16* __restrict__ y1) {
  __shared__ _Float16 xs[66 * 136];
  const int b = blockIdx.y, l0 = blockIdx.x * NT, tid = threadIdx.x;
  for (int task = tid; task < 66 * 16; task += 256) {
    int p = task >> 4, seg = task & 15;
    int l = l0 + p - 1;
    uint4 val = make_uint4(0u, 0u, 0u, 0u);
    if (l >= 0 && l < SEQ) {
      int id = ids[b * SEQ + l];
      val = *(const uint4*)(embF + id * DIM + seg * 8);
    }
    *(uint4*)(&xs[p * 136 + seg * 8]) = val;
  }
  __syncthreads();

  const int wid = tid >> 5, lane = tid & 31;
  const int laneN = lane & 15, laneHi = lane >> 4;
  const v8f z = {0.f, 0.f, 0.f, 0.f, 0.f, 0.f, 0.f, 0.f};
  v8f c[2][4];
#pragma unroll
  for (int mi = 0; mi < 2; ++mi)
#pragma unroll
    for (int nt = 0; nt < 4; ++nt) c[mi][nt] = z;

  const int KC = 12;
  for (int kc = 0; kc < KC; ++kc) {
    int tap = kc >> 2;
    int c0  = ((kc & 3) << 5) + (laneHi << 4);
    Frag a0, a1;
    a0.v = *(const v16h*)(W1p + ((((2 * wid) * KC) + kc) * 32 + lane) * 16);
    a1.v = *(const v16h*)(W1p + ((((2 * wid + 1) * KC) + kc) * 32 + lane) * 16);
#pragma unroll
    for (int nt = 0; nt < 4; ++nt) {
      int p = nt * 16 + laneN + tap;
      Frag bf;
      const uint4* q = (const uint4*)(&xs[p * 136 + c0]);
      bf.q[0] = q[0]; bf.q[1] = q[1];
      c[0][nt] = WMMA_F16(a0.v, bf.v, c[0][nt]);
      c[1][nt] = WMMA_F16(a1.v, bf.v, c[1][nt]);
    }
  }
#pragma unroll
  for (int mi = 0; mi < 2; ++mi) {
    int mcol = (2 * wid + mi) * 16 + laneHi * 8;
#pragma unroll
    for (int nt = 0; nt < 4; ++nt) {
      int row = b * SEQ + l0 + nt * 16 + laneN;
      Half8 o;
#pragma unroll
      for (int r = 0; r < 8; ++r) o.h[r] = (_Float16)c[mi][nt][r];
      *(uint4*)(y1 + row * 256 + mcol) = o.q;
    }
  }
}

// ---------------------------------------------------------------- conv2: BN1+ReLU staged, O=128, Cin=256
__global__ __launch_bounds__(256) void conv2_kernel(const _Float16* __restrict__ y1,
                                                    const float* __restrict__ sc1,
                                                    const float* __restrict__ sh1,
                                                    const _Float16* __restrict__ W2p,
                                                    _Float16* __restrict__ y2) {
  __shared__ _Float16 xs[66 * 264];
  const int b = blockIdx.y, l0 = blockIdx.x * NT, tid = threadIdx.x;
  for (int task = tid; task < 66 * 32; task += 256) {
    int p = task >> 5, seg = task & 31;
    int l = l0 + p - 1;
    uint4 val = make_uint4(0u, 0u, 0u, 0u);
    if (l >= 0 && l < SEQ) {
      Half8 raw, o;
      raw.q = *(const uint4*)(y1 + (b * SEQ + l) * 256 + seg * 8);
#pragma unroll
      for (int e = 0; e < 8; ++e) {
        int ch = seg * 8 + e;
        float v = (float)raw.h[e] * sc1[ch] + sh1[ch];
        o.h[e] = (_Float16)fmaxf(v, 0.f);
      }
      val = o.q;
    }
    *(uint4*)(&xs[p * 264 + seg * 8]) = val;
  }
  __syncthreads();

  const int wid = tid >> 5, lane = tid & 31;
  const int laneN = lane & 15, laneHi = lane >> 4;
  const v8f z = {0.f, 0.f, 0.f, 0.f, 0.f, 0.f, 0.f, 0.f};
  v8f c[4] = {z, z, z, z};
  const int KC = 24;
  for (int kc = 0; kc < KC; ++kc) {
    int tap = kc >> 3;
    int c0  = ((kc & 7) << 5) + (laneHi << 4);
    Frag a;
    a.v = *(const v16h*)(W2p + ((wid * KC + kc) * 32 + lane) * 16);
#pragma unroll
    for (int nt = 0; nt < 4; ++nt) {
      int p = nt * 16 + laneN + tap;
      Frag bf;
      const uint4* q = (const uint4*)(&xs[p * 264 + c0]);
      bf.q[0] = q[0]; bf.q[1] = q[1];
      c[nt] = WMMA_F16(a.v, bf.v, c[nt]);
    }
  }
  int mcol = wid * 16 + laneHi * 8;
#pragma unroll
  for (int nt = 0; nt < 4; ++nt) {
    int row = b * SEQ + l0 + nt * 16 + laneN;
    Half8 o;
#pragma unroll
    for (int r = 0; r < 8; ++r) o.h[r] = (_Float16)c[nt][r];
    *(uint4*)(y2 + row * DIM + mcol) = o.q;
  }
}

// ---------------------------------------------------------------- conv3: BN2+ReLU staged, O=128, Cin=128, bias+ReLU+mean
__global__ __launch_bounds__(256) void conv3_kernel(const _Float16* __restrict__ y2,
                                                    const float* __restrict__ sc2,
                                                    const float* __restrict__ sh2,
                                                    const _Float16* __restrict__ W3p,
                                                    const float* __restrict__ b3,
                                                    float* __restrict__ hcnn) {
  __shared__ _Float16 xs[66 * 136];
  const int b = blockIdx.y, l0 = blockIdx.x * NT, tid = threadIdx.x;
  for (int task = tid; task < 66 * 16; task += 256) {
    int p = task >> 4, seg = task & 15;
    int l = l0 + p - 1;
    uint4 val = make_uint4(0u, 0u, 0u, 0u);
    if (l >= 0 && l < SEQ) {
      Half8 raw, o;
      raw.q = *(const uint4*)(y2 + (b * SEQ + l) * DIM + seg * 8);
#pragma unroll
      for (int e = 0; e < 8; ++e) {
        int ch = seg * 8 + e;
        float v = (float)raw.h[e] * sc2[ch] + sh2[ch];
        o.h[e] = (_Float16)fmaxf(v, 0.f);
      }
      val = o.q;
    }
    *(uint4*)(&xs[p * 136 + seg * 8]) = val;
  }
  __syncthreads();

  const int wid = tid >> 5, lane = tid & 31;
  const int laneN = lane & 15, laneHi = lane >> 4;
  const v8f z = {0.f, 0.f, 0.f, 0.f, 0.f, 0.f, 0.f, 0.f};
  v8f c[4] = {z, z, z, z};
  const int KC = 12;
  for (int kc = 0; kc < KC; ++kc) {
    int tap = kc >> 2;
    int c0  = ((kc & 3) << 5) + (laneHi << 4);
    Frag a;
    a.v = *(const v16h*)(W3p + ((wid * KC + kc) * 32 + lane) * 16);
#pragma unroll
    for (int nt = 0; nt < 4; ++nt) {
      int p = nt * 16 + laneN + tap;
      Frag bf;
      const uint4* q = (const uint4*)(&xs[p * 136 + c0]);
      bf.q[0] = q[0]; bf.q[1] = q[1];
      c[nt] = WMMA_F16(a.v, bf.v, c[nt]);
    }
  }
  // bias + relu + sum over 64 positions per channel, then atomic into hcnn
  float sacc[8];
#pragma unroll
  for (int r = 0; r < 8; ++r) {
    float bias = b3[wid * 16 + laneHi * 8 + r];
    float s = 0.f;
#pragma unroll
    for (int nt = 0; nt < 4; ++nt) s += fmaxf(c[nt][r] + bias, 0.f);
    sacc[r] = s;
  }
#pragma unroll
  for (int m = 1; m <= 8; m <<= 1)
#pragma unroll
    for (int r = 0; r < 8; ++r) sacc[r] += __shfl_xor(sacc[r], m, 32);
  if (laneN == 0) {
#pragma unroll
    for (int r = 0; r < 8; ++r)
      atomicAdd(&hcnn[b * DIM + wid * 16 + laneHi * 8 + r], sacc[r]);
  }
}

// ---------------------------------------------------------------- BN stats over y f16 [rows][O]
__global__ __launch_bounds__(256) void bn_stats_kernel(const _Float16* __restrict__ y,
                                                       float* __restrict__ sum,
                                                       float* __restrict__ sq,
                                                       int O, int rowsPerBlock) {
  int t = threadIdx.x;
  int chan = t & (O - 1);
  int rstep = 256 / O;
  int r0 = blockIdx.x * rowsPerBlock + (t / O);
  int rend = (blockIdx.x + 1) * rowsPerBlock;
  float s = 0.f, q = 0.f;
  for (int r = r0; r < rend; r += rstep) {
    float v = (float)y[r * O + chan];
    s += v; q += v * v;
  }
  atomicAdd(&sum[chan], s);
  atomicAdd(&sq[chan], q);
}

__global__ void bn_final_kernel(const float* __restrict__ sum, const float* __restrict__ sq,
                                const float* __restrict__ g, const float* __restrict__ be,
                                float* __restrict__ sc, float* __restrict__ sh, int O) {
  int i = blockIdx.x * 256 + threadIdx.x;
  if (i >= O) return;
  const float inv = 1.0f / (float)(BATCH * SEQ);
  float m = sum[i] * inv;
  float v = sq[i] * inv - m * m;
  v = fmaxf(v, 0.f);
  float s = g[i] / sqrtf(v + 1e-5f);
  sc[i] = s;
  sh[i] = be[i] - m * s;
}

// ---------------------------------------------------------------- social stats + sws
__global__ __launch_bounds__(256) void social_kernel(const int* __restrict__ ids,
                                                     const float* __restrict__ base,
                                                     float* __restrict__ stats,
                                                     float* __restrict__ swsOut) {
  __shared__ int scnt[256], sseg[256], sfirst[256], slast[256];
  int b = blockIdx.x, t = threadIdx.x;
  int cnt = 0, seg = 0, first = SEQ, last = -1;
  for (int l = t; l < SEQ; l += 256) {
    int id = ids[b * SEQ + l];
    bool m = (id == 1) || (id == 18);
    bool pm = false;
    if (l > 0) {
      int ip = ids[b * SEQ + l - 1];
      pm = (ip == 1) || (ip == 18);
    }
    cnt += m ? 1 : 0;
    seg += (m && !pm) ? 1 : 0;
    if (m) { first = min(first, l); last = max(last, l); }
  }
  scnt[t] = cnt; sseg[t] = seg; sfirst[t] = first; slast[t] = last;
  __syncthreads();
  for (int off = 128; off > 0; off >>= 1) {
    if (t < off) {
      scnt[t] += scnt[t + off];
      sseg[t] += sseg[t + off];
      sfirst[t] = min(sfirst[t], sfirst[t + off]);
      slast[t]  = max(slast[t], slast[t + off]);
    }
    __syncthreads();
  }
  if (t == 0) {
    float total = (float)scnt[0];
    float nseg  = (float)sseg[0];
    float mu    = nseg > 0.f ? total / fmaxf(nseg, 1.f) : 0.f;
    float gaps  = nseg > 0.f ? (float)(slast[0] - sfirst[0] + 1) - total : 0.f;
    float resp  = nseg > 1.f ? gaps / fmaxf(nseg - 1.f, 1.f) : 0.f;
    stats[b * 3 + 0] = mu;
    stats[b * 3 + 1] = nseg;
    stats[b * 3 + 2] = resp;
    float den = base[b * 3 + 0] * base[b * 3 + 1] + 1e-10f;
    float sws = 1.f - (mu * nseg) / den;
    swsOut[b] = fminf(fmaxf(sws, 0.f), 1.f);
  }
}

// ---------------------------------------------------------------- final small GEMM
__global__ __launch_bounds__(128) void final_kernel(const float* __restrict__ hcnn,
                                                    const float* __restrict__ stats,
                                                    const float* __restrict__ Ws,
                                                    const float* __restrict__ bs,
                                                    const float* __restrict__ Wo,
                                                    const float* __restrict__ bo,
                                                    float* __restrict__ out) {
  __shared__ float h[256];
  int b = blockIdx.x, i = threadIdx.x;
  float st0 = stats[b * 3 + 0], st1 = stats[b * 3 + 1], st2 = stats[b * 3 + 2];
  h[i] = hcnn[b * DIM + i] * (1.0f / (float)SEQ);
  h[128 + i] = Ws[i * 3 + 0] * st0 + Ws[i * 3 + 1] * st1 + Ws[i * 3 + 2] * st2 + bs[i];
  __syncthreads();
  float acc = bo[i];
  for (int j = 0; j < 256; ++j) acc += Wo[i * 256 + j] * h[j];
  out[b * DIM + i] = acc;
}

// ---------------------------------------------------------------- host
static const size_t EMB_OFF  = 0;         // 21*128 f16
static const size_t W1P_OFF  = 8192;      // 256*384 f16
static const size_t W2P_OFF  = 204800;    // 128*768 f16
static const size_t W3P_OFF  = 401408;    // 128*384 f16
static const size_t SUM1_OFF = 499712;    // 256 f32
static const size_t SQ1_OFF  = 500736;    // 256 f32 (contiguous with SUM1)
static const size_t SC1_OFF  = 501760;
static const size_t SH1_OFF  = 502784;
static const size_t SUM2_OFF = 503808;    // 128 f32
static const size_t SQ2_OFF  = 504320;    // contiguous with SUM2
static const size_t SC2_OFF  = 504832;
static const size_t SH2_OFF  = 505344;
static const size_t STATS_OFF= 505856;    // 32*3 f32
static const size_t HCNN_OFF = 506368;    // 32*128 f32
static const size_t Y1_OFF   = 524288;                   // 32*8192*256 f16 = 134 MB
static const size_t Y2_OFF   = Y1_OFF + 134217728ull;    // 32*8192*128 f16 = 67 MB

extern "C" void kernel_launch(void* const* d_in, const int* in_sizes, int n_in,
                              void* d_out, int out_size, void* d_ws, size_t ws_size,
                              hipStream_t stream) {
  (void)in_sizes; (void)n_in; (void)out_size; (void)ws_size;
  const int*   ids   = (const int*)d_in[0];
  const float* base  = (const float*)d_in[1];
  const float* embed = (const float*)d_in[2];
  const float* W1    = (const float*)d_in[3];
  const float* g1    = (const float*)d_in[5];
  const float* be1   = (const float*)d_in[6];
  const float* W2    = (const float*)d_in[7];
  const float* g2    = (const float*)d_in[9];
  const float* be2   = (const float*)d_in[10];
  const float* W3    = (const float*)d_in[11];
  const float* b3    = (const float*)d_in[12];
  const float* Ws    = (const float*)d_in[13];
  const float* bs    = (const float*)d_in[14];
  const float* Wo    = (const float*)d_in[15];
  const float* bo    = (const float*)d_in[16];
  float* out = (float*)d_out;
  char*  ws  = (char*)d_ws;

  _Float16* embF = (_Float16*)(ws + EMB_OFF);
  _Float16* W1p  = (_Float16*)(ws + W1P_OFF);
  _Float16* W2p  = (_Float16*)(ws + W2P_OFF);
  _Float16* W3p  = (_Float16*)(ws + W3P_OFF);
  float* sum1 = (float*)(ws + SUM1_OFF);
  float* sq1  = (float*)(ws + SQ1_OFF);
  float* sc1  = (float*)(ws + SC1_OFF);
  float* sh1  = (float*)(ws + SH1_OFF);
  float* sum2 = (float*)(ws + SUM2_OFF);
  float* sq2  = (float*)(ws + SQ2_OFF);
  float* sc2  = (float*)(ws + SC2_OFF);
  float* sh2  = (float*)(ws + SH2_OFF);
  float* stats= (float*)(ws + STATS_OFF);
  float* hcnn = (float*)(ws + HCNN_OFF);
  _Float16* y1 = (_Float16*)(ws + Y1_OFF);
  _Float16* y2 = (_Float16*)(ws + Y2_OFF);

  // zero accumulators (sum1+sq1 contiguous = 512 f32; sum2+sq2 = 256 f32)
  zero_kernel<<<2, 256, 0, stream>>>(sum1, 512);
  zero_kernel<<<1, 256, 0, stream>>>(sum2, 256);
  zero_kernel<<<16, 256, 0, stream>>>(hcnn, BATCH * DIM);

  // weight / embedding preparation
  cast_f16_kernel<<<(21 * DIM + 255) / 256, 256, 0, stream>>>(embed, embF, 21 * DIM);
  pack_w_kernel<<<(256 * 384 + 255) / 256, 256, 0, stream>>>(W1, W1p, 256, 128);
  pack_w_kernel<<<(128 * 768 + 255) / 256, 256, 0, stream>>>(W2, W2p, 128, 256);
  pack_w_kernel<<<(128 * 384 + 255) / 256, 256, 0, stream>>>(W3, W3p, 128, 128);

  // social statistics branch
  social_kernel<<<BATCH, 256, 0, stream>>>(ids, base, stats, out + BATCH * DIM);

  dim3 cgrid(SEQ / NT, BATCH);
  conv1_kernel<<<cgrid, 256, 0, stream>>>(ids, embF, W1p, y1);
  bn_stats_kernel<<<128, 256, 0, stream>>>(y1, sum1, sq1, 256, 2048);
  bn_final_kernel<<<1, 256, 0, stream>>>(sum1, sq1, g1, be1, sc1, sh1, 256);

  conv2_kernel<<<cgrid, 256, 0, stream>>>(y1, sc1, sh1, W2p, y2);
  bn_stats_kernel<<<128, 256, 0, stream>>>(y2, sum2, sq2, 128, 2048);
  bn_final_kernel<<<1, 256, 0, stream>>>(sum2, sq2, g2, be2, sc2, sh2, 128);

  conv3_kernel<<<cgrid, 256, 0, stream>>>(y2, sc2, sh2, W3p, b3, hcnn);

  final_kernel<<<BATCH, 128, 0, stream>>>(hcnn, stats, Ws, bs, Wo, bo, out);
}
